// InputFeedMEMDecoder_22746146799883
// MI455X (gfx1250) — matrix operand order repaired and checked by reading, not compiled, not run on hardware
//
#include <hip/hip_runtime.h>
#include <hip/hip_bf16.h>
#include <math.h>

// Problem dims (fixed by reference)
#define TT 64
#define BB 32
#define SS 128
#define HH 1024
#define EE 1024

typedef __attribute__((ext_vector_type(16))) _Float16 v16h;
typedef __attribute__((ext_vector_type(8)))  _Float16 v8h;
typedef __attribute__((ext_vector_type(8)))  float    v8f;
typedef __attribute__((ext_vector_type(4)))  float    v4f;

__device__ __forceinline__ float sigmoidf_(float x) { return 1.0f / (1.0f + __expf(-x)); }

// ---------------------------------------------------------------------------
// WMMA fragment helpers (same contiguous k order for A and B)
//   A 16x32 f16: lanes 0-15 row m, K {k+0..15}; lanes 16-31 K {k+16..31}
//   B 32x16 f16 (W^T): lanes 0-15 col n, K {k+0..15}; lanes 16-31 K {k+16..31}
// ---------------------------------------------------------------------------
__device__ __forceinline__ v16h load_a_f16(const _Float16* __restrict__ row,
                                           int akoff0, int akoff1)
{
    v8h lo = *(const v8h*)(row + akoff0);
    v8h hi = *(const v8h*)(row + akoff1);
    v16h a;
#pragma unroll
    for (int i = 0; i < 8; ++i) { a[i] = lo[i]; a[i + 8] = hi[i]; }
    return a;
}

__device__ __forceinline__ v16h load_a_f32(const float* __restrict__ row,
                                           int akoff0, int akoff1)
{
    v4f f0 = *(const v4f*)(row + akoff0);
    v4f f1 = *(const v4f*)(row + akoff0 + 4);
    v4f f2 = *(const v4f*)(row + akoff1);
    v4f f3 = *(const v4f*)(row + akoff1 + 4);
    v16h a;
#pragma unroll
    for (int i = 0; i < 4; ++i) {
        a[i]      = (_Float16)f0[i];
        a[i + 4]  = (_Float16)f1[i];
        a[i + 8]  = (_Float16)f2[i];
        a[i + 12] = (_Float16)f3[i];
    }
    return a;
}

// wp already points at this lane's W row + bkoff + k
__device__ __forceinline__ v8f wmma_acc(v16h af, const _Float16* __restrict__ wp, v8f acc)
{
    v8h blo = *(const v8h*)(wp);
    v8h bhi = *(const v8h*)(wp + 8);
    v16h bf;
#pragma unroll
    for (int i = 0; i < 8; ++i) { bf[i] = blo[i]; bf[i + 8] = bhi[i]; }
    return __builtin_amdgcn_wmma_f32_16x16x32_f16(
        false, af, false, bf, (short)0, acc, false, false);
}

// ---------------------------------------------------------------------------
// Generic WMMA GEMM:  C[M,N](f32) = A[M,K](f16,lda) * W[N,K]^T(f16,ldw)
// Optional bias[n], tanh epilogue, secondary f16 output C16 (ldc16).
// One wave -> one 16x16 C tile. K%32==0, N%16==0, M%16==0.
// ---------------------------------------------------------------------------
__global__ __launch_bounds__(128)
void gemm_wmma_kernel(const _Float16* __restrict__ A, int lda,
                      const _Float16* __restrict__ W, int ldw,
                      float* __restrict__ C, int ldc,
                      const float* __restrict__ bias,
                      _Float16* __restrict__ C16, int ldc16,
                      int M, int N, int K, int do_tanh)
{
    const int wave   = blockIdx.x * (blockDim.x >> 5) + (threadIdx.x >> 5);
    const int lane   = threadIdx.x & 31;
    const int ntiles = N >> 4;
    const int tile_n = wave % ntiles;
    const int tile_m = wave / ntiles;
    if (tile_m * 16 >= M) return;

    const int half = lane >> 4;
    const int l16  = lane & 15;
    const int akoff0 = half ? 16 : 0, akoff1 = half ? 24 : 8;
    const int bkoff  = half * 16;

    const _Float16* arow = A + (size_t)(tile_m * 16 + l16) * (size_t)lda;
    const _Float16* wrow = W + (size_t)(tile_n * 16 + l16) * (size_t)ldw + bkoff;

    v8f acc = {};
    for (int k = 0; k < K; k += 32) {
        v16h af = load_a_f16(arow + k, akoff0, akoff1);
        acc = wmma_acc(af, wrow + k, acc);
    }

    const int n  = tile_n * 16 + l16;
    const float bv = bias ? bias[n] : 0.0f;
#pragma unroll
    for (int i = 0; i < 8; ++i) {
        const int m = tile_m * 16 + half * 8 + i;
        float v = acc[i] + bv;
        if (do_tanh) v = tanhf(v);
        C[(size_t)m * (size_t)ldc + n] = v;
        if (C16) C16[(size_t)m * (size_t)ldc16 + n] = (_Float16)v;
    }
}

// ---------------------------------------------------------------------------
// Fused LSTM epilogue: gates -> c update -> h (f16 x2 destinations)
// ---------------------------------------------------------------------------
__device__ __forceinline__ void lstm_epilogue(v8f gi, v8f gf, v8f gg, v8f go,
                                              const float* __restrict__ bias,
                                              int n, int mbase,
                                              float* __restrict__ cstate,
                                              _Float16* __restrict__ hdst0, int ld0,
                                              _Float16* __restrict__ hdst1, int ld1)
{
    const float bi = bias[n];
    const float bf_ = bias[HH + n];
    const float bg = bias[2 * HH + n];
    const float bo = bias[3 * HH + n];
#pragma unroll
    for (int i = 0; i < 8; ++i) {
        const int m = mbase + i;
        const size_t ci = (size_t)m * HH + n;
        float iv = sigmoidf_(gi[i] + bi);
        float fv = sigmoidf_(gf[i] + bf_);
        float gv = tanhf(gg[i] + bg);
        float ov = sigmoidf_(go[i] + bo);
        float cn = fv * cstate[ci] + iv * gv;
        cstate[ci] = cn;
        float hn = ov * tanhf(cn);
        hdst0[(size_t)m * ld0 + n] = (_Float16)hn;
        hdst1[(size_t)m * ld1 + n] = (_Float16)hn;
    }
}

// ---------------------------------------------------------------------------
// Layer-0 LSTM, fully fused: A = [emb(gathered fp32)|feed|h0], K=3072.
// Each wave: 16 batch rows x 16 hidden cols across all 4 gates (4 WMMA accums).
// 128 waves total (64 n-tiles x 2 m-tiles).
// ---------------------------------------------------------------------------
__global__ __launch_bounds__(128)
void lstm0_fused_kernel(const int* __restrict__ tgt, int t,
                        const float* __restrict__ emb_table,
                        const _Float16* __restrict__ X0,   // [B,2048] f16 = [feed|h0]
                        const _Float16* __restrict__ W,    // [4096,3072] f16
                        const float* __restrict__ bias,    // [4096]
                        float* __restrict__ cstate,        // [B,H]
                        _Float16* __restrict__ hdst0, int ld0,
                        _Float16* __restrict__ hdst1, int ld1)
{
    const int wave = blockIdx.x * (blockDim.x >> 5) + (threadIdx.x >> 5);
    const int lane = threadIdx.x & 31;
    const int tile_n = wave & 63;
    const int tile_m = wave >> 6;          // 0..1
    const int half = lane >> 4, l16 = lane & 15;
    const int akoff0 = half ? 16 : 0, akoff1 = half ? 24 : 8;
    const int bkoff  = half * 16;
    const int am = tile_m * 16 + l16;      // batch row for A fragment
    const int n  = tile_n * 16 + l16;      // hidden column

    const float*    erow = emb_table + (size_t)tgt[t * BB + am] * EE;
    const _Float16* xrow = X0 + (size_t)am * 2048;
    const _Float16* w0 = W + (size_t)(0 * HH + n) * 3072 + bkoff;
    const _Float16* w1 = W + (size_t)(1 * HH + n) * 3072 + bkoff;
    const _Float16* w2 = W + (size_t)(2 * HH + n) * 3072 + bkoff;
    const _Float16* w3 = W + (size_t)(3 * HH + n) * 3072 + bkoff;

    v8f a0 = {}, a1 = {}, a2 = {}, a3 = {};
    // K segment 0..1023: embedding rows (fp32 gather, convert in-register)
    for (int k = 0; k < 1024; k += 32) {
        v16h af = load_a_f32(erow + k, akoff0, akoff1);
        a0 = wmma_acc(af, w0 + k, a0);
        a1 = wmma_acc(af, w1 + k, a1);
        a2 = wmma_acc(af, w2 + k, a2);
        a3 = wmma_acc(af, w3 + k, a3);
    }
    // K segment 1024..3071: [feed|h0] f16 state
    for (int k = 0; k < 2048; k += 32) {
        v16h af = load_a_f16(xrow + k, akoff0, akoff1);
        a0 = wmma_acc(af, w0 + 1024 + k, a0);
        a1 = wmma_acc(af, w1 + 1024 + k, a1);
        a2 = wmma_acc(af, w2 + 1024 + k, a2);
        a3 = wmma_acc(af, w3 + 1024 + k, a3);
    }
    lstm_epilogue(a0, a1, a2, a3, bias, n, tile_m * 16 + half * 8,
                  cstate, hdst0, ld0, hdst1, ld1);
}

// ---------------------------------------------------------------------------
// Layer-1 LSTM, fused: A = [h0n|h1] f16, K=2048.
// ---------------------------------------------------------------------------
__global__ __launch_bounds__(128)
void lstm1_fused_kernel(const _Float16* __restrict__ X1,   // [B,2048] f16 = [h0n|h1]
                        const _Float16* __restrict__ W,    // [4096,2048] f16
                        const float* __restrict__ bias,
                        float* __restrict__ cstate,
                        _Float16* __restrict__ hdst0, int ld0,
                        _Float16* __restrict__ hdst1, int ld1)
{
    const int wave = blockIdx.x * (blockDim.x >> 5) + (threadIdx.x >> 5);
    const int lane = threadIdx.x & 31;
    const int tile_n = wave & 63;
    const int tile_m = wave >> 6;
    const int half = lane >> 4, l16 = lane & 15;
    const int akoff0 = half ? 16 : 0, akoff1 = half ? 24 : 8;
    const int bkoff  = half * 16;
    const int am = tile_m * 16 + l16;
    const int n  = tile_n * 16 + l16;

    const _Float16* xrow = X1 + (size_t)am * 2048;
    const _Float16* w0 = W + (size_t)(0 * HH + n) * 2048 + bkoff;
    const _Float16* w1 = W + (size_t)(1 * HH + n) * 2048 + bkoff;
    const _Float16* w2 = W + (size_t)(2 * HH + n) * 2048 + bkoff;
    const _Float16* w3 = W + (size_t)(3 * HH + n) * 2048 + bkoff;

    v8f a0 = {}, a1 = {}, a2 = {}, a3 = {};
    for (int k = 0; k < 2048; k += 32) {
        v16h af = load_a_f16(xrow + k, akoff0, akoff1);
        a0 = wmma_acc(af, w0 + k, a0);
        a1 = wmma_acc(af, w1 + k, a1);
        a2 = wmma_acc(af, w2 + k, a2);
        a3 = wmma_acc(af, w3 + k, a3);
    }
    lstm_epilogue(a0, a1, a2, a3, bias, n, tile_m * 16 + half * 8,
                  cstate, hdst0, ld0, hdst1, ld1);
}

// ---------------------------------------------------------------------------
// Fused attention: score + softmax + context. One 256-thread WG per batch b.
// ---------------------------------------------------------------------------
__global__ __launch_bounds__(256)
void attn_fused_kernel(const float* __restrict__ q,        // [B,H]
                       const _Float16* __restrict__ mem,   // [B,S,H] f16
                       float* __restrict__ attns_t,        // [B,S] (d_out region)
                       _Float16* __restrict__ ctx)         // concat2, ld 2048
{
    const int b = blockIdx.x;
    const int tid = threadIdx.x;
    const int wave = tid >> 5, lane = tid & 31;
    __shared__ float s_align[SS];

    const _Float16* mb = mem + (size_t)b * SS * HH;
    const float*    qb = q + (size_t)b * HH;

    // scores: each of 8 waves handles 16 source positions
    for (int s = wave; s < SS; s += 8) {
        const _Float16* mr = mb + (size_t)s * HH;
        float acc = 0.0f;
        for (int h = lane; h < HH; h += 32) acc += qb[h] * (float)mr[h];
        for (int off = 16; off; off >>= 1) acc += __shfl_xor(acc, off, 32);
        if (lane == 0) s_align[s] = acc;
    }
    __syncthreads();

    // softmax over S=128, wave 0 only
    if (wave == 0) {
        float v[4];
        float mx = -3.0e38f;
#pragma unroll
        for (int i = 0; i < 4; ++i) { v[i] = s_align[i * 32 + lane]; mx = fmaxf(mx, v[i]); }
        for (int off = 16; off; off >>= 1) mx = fmaxf(mx, __shfl_xor(mx, off, 32));
        float sum = 0.0f;
#pragma unroll
        for (int i = 0; i < 4; ++i) { v[i] = __expf(v[i] - mx); sum += v[i]; }
        for (int off = 16; off; off >>= 1) sum += __shfl_xor(sum, off, 32);
        float inv = 1.0f / sum;
#pragma unroll
        for (int i = 0; i < 4; ++i) {
            float a = v[i] * inv;
            s_align[i * 32 + lane] = a;
            attns_t[b * SS + i * 32 + lane] = a;
        }
    }
    __syncthreads();

    // context: 4 hidden elements per thread
    for (int h = tid; h < HH; h += 256) {
        float acc = 0.0f;
        for (int s = 0; s < SS; ++s) acc += s_align[s] * (float)mb[(size_t)s * HH + h];
        ctx[(size_t)b * 2048 + h] = (_Float16)acc;
    }
}

// ---------------------------------------------------------------------------
// One-time prep kernels
// ---------------------------------------------------------------------------
__global__ void cat_convert_kernel(const float* __restrict__ A, int ka,
                                   const float* __restrict__ B, int kb,
                                   _Float16* __restrict__ dst, int nrows)
{
    const int kt = ka + kb;
    size_t idx = (size_t)blockIdx.x * blockDim.x + threadIdx.x;
    if (idx >= (size_t)nrows * kt) return;
    int r = (int)(idx / kt);
    int c = (int)(idx % kt);
    float v = (c < ka) ? A[(size_t)r * ka + c] : B[(size_t)r * kb + (c - ka)];
    dst[idx] = (_Float16)v;
}

__global__ void convert_kernel(const float* __restrict__ src,
                               _Float16* __restrict__ dst, size_t n)
{
    size_t idx = (size_t)blockIdx.x * blockDim.x + threadIdx.x;
    if (idx < n) dst[idx] = (_Float16)src[idx];
}

// memory_bank [S,B,H] f32 -> mem16 [B,S,H] f16
__global__ void mem_transpose_kernel(const float* __restrict__ mb,
                                     _Float16* __restrict__ dst)
{
    size_t idx = (size_t)blockIdx.x * blockDim.x + threadIdx.x;
    if (idx >= (size_t)BB * SS * HH) return;
    int h = (int)(idx & (HH - 1));
    int s = (int)((idx >> 10) & (SS - 1));
    int b = (int)(idx >> 17);
    dst[idx] = (_Float16)mb[((size_t)s * BB + b) * HH + h];
}

__global__ void bias_sum_kernel(const float* __restrict__ a,
                                const float* __restrict__ b,
                                float* __restrict__ dst, int n)
{
    int idx = blockIdx.x * blockDim.x + threadIdx.x;
    if (idx < n) dst[idx] = a[idx] + b[idx];
}

__global__ void init_state_kernel(const float* __restrict__ input_feed,
                                  const float* __restrict__ h0,
                                  const float* __restrict__ c0,
                                  _Float16* __restrict__ X0,   // buffer 0: [feed|h0]
                                  _Float16* __restrict__ X1,   // buffer 0: [h0n|h1]
                                  float* __restrict__ c0s, float* __restrict__ c1s)
{
    int idx = blockIdx.x * blockDim.x + threadIdx.x;   // B*H
    if (idx >= BB * HH) return;
    int b = idx >> 10, h = idx & (HH - 1);
    X0[(size_t)b * 2048 + h]        = (_Float16)input_feed[idx];    // feed
    X0[(size_t)b * 2048 + 1024 + h] = (_Float16)h0[idx];            // h layer0
    X1[(size_t)b * 2048 + 1024 + h] = (_Float16)h0[BB * HH + idx];  // h layer1
    c0s[idx] = c0[idx];
    c1s[idx] = c0[BB * HH + idx];
}

// ---------------------------------------------------------------------------
extern "C" void kernel_launch(void* const* d_in, const int* in_sizes, int n_in,
                              void* d_out, int out_size, void* d_ws, size_t ws_size,
                              hipStream_t stream)
{
    const int*   tgt         = (const int*)  d_in[0];
    const float* memory_bank = (const float*)d_in[1];
    const float* input_feed  = (const float*)d_in[2];
    const float* h0          = (const float*)d_in[3];
    const float* c0          = (const float*)d_in[4];
    const float* emb_table   = (const float*)d_in[5];
    const float* W_ih0       = (const float*)d_in[6];
    const float* b_ih0       = (const float*)d_in[7];
    const float* W_hh0       = (const float*)d_in[8];
    const float* b_hh0       = (const float*)d_in[9];
    const float* W_ih1       = (const float*)d_in[10];
    const float* b_ih1       = (const float*)d_in[11];
    const float* W_hh1       = (const float*)d_in[12];
    const float* b_hh1       = (const float*)d_in[13];
    const float* W_in        = (const float*)d_in[14];
    const float* W_out       = (const float*)d_in[15];

    float* outs  = (float*)d_out;                    // [T,B,H]
    float* attns = outs + (size_t)TT * BB * HH;      // [T,B,S]

    // ---- workspace carve ----
    char* wp = (char*)d_ws;
    auto carve = [&](size_t bytes) -> char* {
        char* p = wp;
        wp += (bytes + 255) & ~(size_t)255;
        return p;
    };
    _Float16* Wcat0   = (_Float16*)carve((size_t)4096 * 3072 * 2);  // [W_ih0|W_hh0]
    _Float16* Wcat1   = (_Float16*)carve((size_t)4096 * 2048 * 2);  // [W_ih1|W_hh1]
    _Float16* Win16   = (_Float16*)carve((size_t)1024 * 1024 * 2);
    _Float16* Wout16  = (_Float16*)carve((size_t)1024 * 2048 * 2);
    _Float16* mem16   = (_Float16*)carve((size_t)BB * SS * HH * 2);
    _Float16* X0buf[2], *X1buf[2];
    X0buf[0] = (_Float16*)carve((size_t)BB * 2048 * 2);  // [feed|h0] ping
    X0buf[1] = (_Float16*)carve((size_t)BB * 2048 * 2);  //           pong
    X1buf[0] = (_Float16*)carve((size_t)BB * 2048 * 2);  // [h0n|h1]  ping
    X1buf[1] = (_Float16*)carve((size_t)BB * 2048 * 2);  //           pong
    _Float16* concat2 = (_Float16*)carve((size_t)BB * 2048 * 2);     // [ctx|h1n]
    float*    c0s     = (float*)   carve((size_t)BB * HH * 4);
    float*    c1s     = (float*)   carve((size_t)BB * HH * 4);
    float*    qbuf    = (float*)   carve((size_t)BB * HH * 4);
    float*    b0sum   = (float*)   carve((size_t)4096 * 4);
    float*    b1sum   = (float*)   carve((size_t)4096 * 4);
    (void)ws_size; (void)in_sizes; (void)n_in; (void)out_size;

    // ---- one-time prep ----
    cat_convert_kernel<<<(4096u * 3072u + 255) / 256, 256, 0, stream>>>(W_ih0, 2048, W_hh0, 1024, Wcat0, 4096);
    cat_convert_kernel<<<(4096u * 2048u + 255) / 256, 256, 0, stream>>>(W_ih1, 1024, W_hh1, 1024, Wcat1, 4096);
    convert_kernel<<<(1024u * 1024u + 255) / 256, 256, 0, stream>>>(W_in,  Win16,  (size_t)1024 * 1024);
    convert_kernel<<<(1024u * 2048u + 255) / 256, 256, 0, stream>>>(W_out, Wout16, (size_t)1024 * 2048);
    mem_transpose_kernel<<<((unsigned)(BB * SS * HH) + 255) / 256, 256, 0, stream>>>(memory_bank, mem16);
    bias_sum_kernel<<<16, 256, 0, stream>>>(b_ih0, b_hh0, b0sum, 4096);
    bias_sum_kernel<<<16, 256, 0, stream>>>(b_ih1, b_hh1, b1sum, 4096);
    init_state_kernel<<<(BB * HH + 255) / 256, 256, 0, stream>>>(
        input_feed, h0, c0, X0buf[0], X1buf[0], c0s, c1s);

    // ---- sequential decode loop: 5 kernels per step ----
    for (int t = 0; t < TT; ++t) {
        const int p  = t & 1;       // read buffers
        const int np = p ^ 1;       // write buffers (next step's read)
        float* outs_t  = outs  + (size_t)t * BB * HH;
        float* attns_t = attns + (size_t)t * BB * SS;

        // layer 0: gates + pointwise fused (M=32, N=4x1024, K=3072); 128 waves
        lstm0_fused_kernel<<<32, 128, 0, stream>>>(
            tgt, t, emb_table, X0buf[p], Wcat0, b0sum, c0s,
            X0buf[np] + 1024, 2048,        // h0(t) -> next step layer-0 input
            X1buf[p], 2048);               // h0(t) -> this step layer-1 input

        // layer 1: fused (M=32, N=4x1024, K=2048); 128 waves
        lstm1_fused_kernel<<<32, 128, 0, stream>>>(
            X1buf[p], Wcat1, b1sum, c1s,
            X1buf[np] + 1024, 2048,        // h1(t) -> next step layer-1 input
            concat2 + 1024, 2048);         // h1(t) -> attention / output concat

        // q = h1n @ W_in^T  (M=32, N=1024, K=1024)
        gemm_wmma_kernel<<<32, 128, 0, stream>>>(
            concat2 + 1024, 2048, Win16, 1024, qbuf, 1024,
            (const float*)nullptr, (_Float16*)nullptr, 0,
            BB, 1024, 1024, 0);

        // attention: score + softmax + context (writes attns + ctx)
        attn_fused_kernel<<<BB, 256, 0, stream>>>(qbuf, mem16, attns_t, concat2);

        // out = tanh([ctx|h1n] @ W_out^T); f32 -> d_out, f16 feed -> next X0
        gemm_wmma_kernel<<<32, 128, 0, stream>>>(
            concat2, 2048, Wout16, 2048, outs_t, 1024,
            (const float*)nullptr, X0buf[np], 2048,
            BB, 1024, 2048, 1);
    }
}
